// FastSelfAttention_29575144801176
// MI455X (gfx1250) — matrix operand order repaired and checked by reading, not compiled, not run on hardware
//
#include <hip/hip_runtime.h>
#include <hip/hip_bf16.h>

#define BATCH 8
#define SEQ   4096
#define HIDN  1024
#define NHEAD 16
#define HDIM  64
#define MTOT  (BATCH*SEQ)
#define RSCALE 0.125f   // 1/sqrt(64)

typedef __attribute__((ext_vector_type(16))) __bf16 bf16x16;
typedef __attribute__((ext_vector_type(4)))  __bf16 bf16x4;
typedef __attribute__((ext_vector_type(8)))  float  f32x8;
typedef __attribute__((ext_vector_type(4)))  float  f32x4;
typedef __attribute__((ext_vector_type(4)))  unsigned int u32x4;

// ---------------------------------------------------------------------------
// Wave-level WMMA fragment loaders (wave32, v_wmma_f32_16x16x32_bf16).
// LDS tiles are row-major [row][32 bf16 K-values].
// A 16x32 bf16 (ISA 7.12.2): lanes 0-15 row M=lane, K={0..7,16..23};
// lanes 16-31 row M=lane-16, K={8..15,24..31} -> two 16B chunks per lane.
// B 32x16 bf16: lanes 0-15 col N=lane K={0..7,16..23}; lanes 16-31 K={8..15,24..31}
// -> two 16B chunks per lane when the tile is stored [N][K].
// ---------------------------------------------------------------------------
__device__ __forceinline__ bf16x16 frag_a(const __bf16* sA, int mbase, int lane) {
  int row = mbase + (lane & 15);
  int kb  = (lane >> 4) * 8;
  union { bf16x16 v; u32x4 q[2]; } f;
  f.q[0] = *(const u32x4*)(sA + row * 32 + kb);
  f.q[1] = *(const u32x4*)(sA + row * 32 + 16 + kb);
  return f.v;
}
__device__ __forceinline__ bf16x16 frag_b(const __bf16* sB, int nbase, int lane) {
  int row = nbase + (lane & 15);
  int kh  = (lane >> 4) * 8;
  union { bf16x16 v; u32x4 q[2]; } f;
  f.q[0] = *(const u32x4*)(sB + row * 32 + kh);
  f.q[1] = *(const u32x4*)(sB + row * 32 + 16 + kh);
  return f.v;
}

// ---------------------------------------------------------------------------
// GEMM: out[m,n] = sum_k A[m,k]*W[n,k] + bias[n]   (A,W f32; compute in bf16)
// Block 256 thr = 8 waves; block tile 128(M) x 256(N) x 32(K); wave tile
// 64x64 (4x4 WMMA). Double-buffered LDS: stage tile k+1 during compute of k.
// ---------------------------------------------------------------------------
template<bool WRITE_BF16>
__global__ __launch_bounds__(256)
void k_gemm_f32(const float* __restrict__ A, const float* __restrict__ W,
                const float* __restrict__ bias,
                float* __restrict__ outF, __bf16* __restrict__ outB) {
  __shared__ __align__(16) __bf16 sA[2][128 * 32];   // 2 x 8 KB
  __shared__ __align__(16) __bf16 sB[2][256 * 32];   // 2 x 16 KB
  const int tid  = threadIdx.x;
  const int lane = tid & 31, wave = tid >> 5;
  const int m0 = blockIdx.y * 128;
  const int n0 = blockIdx.x * 256;
  const int wm = (wave & 1) * 64;
  const int wn = (wave >> 1) * 64;

  const f32x8 vzero = {0.f,0.f,0.f,0.f,0.f,0.f,0.f,0.f};
  f32x8 acc[4][4];
  #pragma unroll
  for (int i = 0; i < 4; ++i)
    #pragma unroll
    for (int j = 0; j < 4; ++j) acc[i][j] = vzero;

  auto stage = [&](int buf, int k0) {
    #pragma unroll
    for (int it = 0; it < 4; ++it) {          // A: 128x32
      int idx = it * 1024 + tid * 4;
      int r = idx >> 5, c = idx & 31;
      f32x4 v = *(const f32x4*)(A + (size_t)(m0 + r) * HIDN + k0 + c);
      bf16x4 bv;
      bv.x = (__bf16)v.x; bv.y = (__bf16)v.y; bv.z = (__bf16)v.z; bv.w = (__bf16)v.w;
      *(bf16x4*)(&sA[buf][r * 32 + c]) = bv;
    }
    #pragma unroll
    for (int it = 0; it < 8; ++it) {          // B: 256x32
      int idx = it * 1024 + tid * 4;
      int r = idx >> 5, c = idx & 31;
      f32x4 v = *(const f32x4*)(W + (size_t)(n0 + r) * HIDN + k0 + c);
      bf16x4 bv;
      bv.x = (__bf16)v.x; bv.y = (__bf16)v.y; bv.z = (__bf16)v.z; bv.w = (__bf16)v.w;
      *(bf16x4*)(&sB[buf][r * 32 + c]) = bv;
    }
  };

  stage(0, 0);
  __syncthreads();
  int buf = 0;
  for (int kt = 0; kt < HIDN / 32; ++kt) {
    if (kt + 1 < HIDN / 32) stage(buf ^ 1, (kt + 1) * 32);   // overlap with WMMA
    bf16x16 bfv[4];
    #pragma unroll
    for (int ni = 0; ni < 4; ++ni) bfv[ni] = frag_b(&sB[buf][0], wn + ni * 16, lane);
    #pragma unroll
    for (int mi = 0; mi < 4; ++mi) {
      bf16x16 af = frag_a(&sA[buf][0], wm + mi * 16, lane);
      #pragma unroll
      for (int ni = 0; ni < 4; ++ni)
        acc[mi][ni] = __builtin_amdgcn_wmma_f32_16x16x32_bf16(
            false, af, false, bfv[ni], (short)0, acc[mi][ni], false, false);
    }
    __syncthreads();
    buf ^= 1;
  }

  // epilogue: C/D layout — VGPR r: lanes 0-15 M=r, lanes 16-31 M=8+r; N=lane&15
  #pragma unroll
  for (int mi = 0; mi < 4; ++mi) {
    #pragma unroll
    for (int ni = 0; ni < 4; ++ni) {
      int n     = n0 + wn + ni * 16 + (lane & 15);
      int mbase = m0 + wm + mi * 16 + ((lane >> 4) << 3);
      float bsv = bias[n];
      #pragma unroll
      for (int r = 0; r < 8; ++r) {
        float v = acc[mi][ni][r] + bsv;
        size_t off = (size_t)(mbase + r) * HIDN + n;
        outF[off] = v;
        if constexpr (WRITE_BF16) outB[off] = (__bf16)v;
      }
    }
  }
}

// ---------------------------------------------------------------------------
// Final GEMM: out[m,n] = sum_k Abf[m,k]*(W[n,k]*gain[b,k]) + bias[n] + resid
// A is bf16 in global -> staged with CDNA5 async global->LDS DMA, double
// buffered: tile k+1 streams in while tile k's WMMAs run; one
// s_wait_asynccnt + barrier per k-step.
// ---------------------------------------------------------------------------
__global__ __launch_bounds__(256)
void k_gemm_final(const __bf16* __restrict__ Abf, const float* __restrict__ W,
                  const float* __restrict__ bias, const float* __restrict__ gain,
                  const float* __restrict__ resid, float* __restrict__ outF) {
  __shared__ __align__(16) __bf16 sA[2][128 * 32];
  __shared__ __align__(16) __bf16 sB[2][256 * 32];
  const int tid  = threadIdx.x;
  const int lane = tid & 31, wave = tid >> 5;
  const int m0 = blockIdx.y * 128;
  const int n0 = blockIdx.x * 256;
  const int wm = (wave & 1) * 64;
  const int wn = (wave >> 1) * 64;
  const int b  = m0 / SEQ;                       // tile lies inside one batch
  const float* g = gain + (size_t)b * HIDN;

  const f32x8 vzero = {0.f,0.f,0.f,0.f,0.f,0.f,0.f,0.f};
  f32x8 acc[4][4];
  #pragma unroll
  for (int i = 0; i < 4; ++i)
    #pragma unroll
    for (int j = 0; j < 4; ++j) acc[i][j] = vzero;

  auto stage = [&](int buf, int k0) {
    // A tile 128x32 bf16 via async DMA: 512 x 16B chunks, 2 per thread
    #pragma unroll
    for (int it = 0; it < 2; ++it) {
      int chunk = it * 256 + tid;
      int r = chunk >> 2, c = (chunk & 3) * 8;
      unsigned ldsoff = (unsigned)(size_t)(&sA[buf][r * 32 + c]);
      unsigned long long ga =
          (unsigned long long)(size_t)(Abf + (size_t)(m0 + r) * HIDN + k0 + c);
      asm volatile("global_load_async_to_lds_b128 %0, %1, off"
                   :: "v"(ldsoff), "v"(ga) : "memory");
    }
    // B tile 256x32 with per-batch gain folded into the bf16 conversion
    #pragma unroll
    for (int it = 0; it < 8; ++it) {
      int idx = it * 1024 + tid * 4;
      int r = idx >> 5, c = idx & 31;
      f32x4 v  = *(const f32x4*)(W + (size_t)(n0 + r) * HIDN + k0 + c);
      f32x4 gv = *(const f32x4*)(g + k0 + c);
      v = v * gv;
      bf16x4 bv;
      bv.x = (__bf16)v.x; bv.y = (__bf16)v.y; bv.z = (__bf16)v.z; bv.w = (__bf16)v.w;
      *(bf16x4*)(&sB[buf][r * 32 + c]) = bv;
    }
  };

  stage(0, 0);
  asm volatile("s_wait_asynccnt 0" ::: "memory");
  __syncthreads();
  int buf = 0;
  for (int kt = 0; kt < HIDN / 32; ++kt) {
    if (kt + 1 < HIDN / 32) stage(buf ^ 1, (kt + 1) * 32);
    bf16x16 bfv[4];
    #pragma unroll
    for (int ni = 0; ni < 4; ++ni) bfv[ni] = frag_b(&sB[buf][0], wn + ni * 16, lane);
    #pragma unroll
    for (int mi = 0; mi < 4; ++mi) {
      bf16x16 af = frag_a(&sA[buf][0], wm + mi * 16, lane);
      #pragma unroll
      for (int ni = 0; ni < 4; ++ni)
        acc[mi][ni] = __builtin_amdgcn_wmma_f32_16x16x32_bf16(
            false, af, false, bfv[ni], (short)0, acc[mi][ni], false, false);
    }
    asm volatile("s_wait_asynccnt 0" ::: "memory");
    __syncthreads();
    buf ^= 1;
  }

  #pragma unroll
  for (int mi = 0; mi < 4; ++mi) {
    #pragma unroll
    for (int ni = 0; ni < 4; ++ni) {
      int n     = n0 + wn + ni * 16 + (lane & 15);
      int mbase = m0 + wm + mi * 16 + ((lane >> 4) << 3);
      float bsv = bias[n];
      #pragma unroll
      for (int r = 0; r < 8; ++r) {
        size_t off = (size_t)(mbase + r) * HIDN + n;
        outF[off] = acc[mi][ni][r] + bsv + resid[off];
      }
    }
  }
}

// ---------------------------------------------------------------------------
// score[b,h,s] = ( sum_j X[b,s,j]*(gain?gain[b,j]:1)*Wh[h,j] + bh[h] )*RSCALE
//                + mask[b,0,s]
// One wave per row; 16-head weights cached in LDS (64 KB).
// ---------------------------------------------------------------------------
__global__ __launch_bounds__(256)
void k_score(const float* __restrict__ X, const float* __restrict__ Wh,
             const float* __restrict__ bh, const float* __restrict__ mask,
             const float* __restrict__ gain, float* __restrict__ score) {
  __shared__ float sW[NHEAD * HIDN];
  const int tid = threadIdx.x;
  for (int i = tid; i < NHEAD * HIDN; i += 256) sW[i] = Wh[i];
  __syncthreads();

  const int lane = tid & 31, wave = tid >> 5;
  const long row = (long)blockIdx.x * 8 + wave;   // b*SEQ + s
  const int b = (int)(row >> 12);                 // SEQ == 4096
  const int s = (int)(row & (SEQ - 1));
  const float* x = X + (size_t)row * HIDN;
  const float* g = gain ? gain + (size_t)b * HIDN : nullptr;

  float acc[NHEAD];
  #pragma unroll
  for (int h = 0; h < NHEAD; ++h) acc[h] = 0.f;

  for (int j0 = 0; j0 < HIDN; j0 += 32) {
    float xv = x[j0 + lane];
    if (g) xv *= g[j0 + lane];
    #pragma unroll
    for (int h = 0; h < NHEAD; ++h) acc[h] += xv * sW[h * HIDN + j0 + lane];
  }
  #pragma unroll
  for (int h = 0; h < NHEAD; ++h) {
    float v = acc[h];
    #pragma unroll
    for (int off = 16; off > 0; off >>= 1) v += __shfl_xor(v, off, 32);
    acc[h] = v;
  }
  if (lane < NHEAD) {
    float sc = (acc[lane] + bh[lane]) * RSCALE + mask[(size_t)b * SEQ + s];
    score[((size_t)b * NHEAD + lane) * SEQ + s] = sc;
  }
}

// ---------------------------------------------------------------------------
// Softmax over S then pooled[b, h*HD+d] = (sum_s w_s * X[b,s,h*HD+d]) * gain?
// One block per (b,h).
// ---------------------------------------------------------------------------
__global__ __launch_bounds__(256)
void k_softpool(const float* __restrict__ score, const float* __restrict__ X,
                const float* __restrict__ gain, float* __restrict__ pooled) {
  __shared__ float se[SEQ];      // 16 KB
  __shared__ float red[256];
  __shared__ float pr[256];
  const int b = blockIdx.x / NHEAD, h = blockIdx.x % NHEAD;
  const int tid = threadIdx.x;
  const float* sc = score + ((size_t)b * NHEAD + h) * SEQ;

  float mx = -__builtin_inff();
  for (int s = tid; s < SEQ; s += 256) { float v = sc[s]; se[s] = v; mx = fmaxf(mx, v); }
  red[tid] = mx; __syncthreads();
  for (int st = 128; st > 0; st >>= 1) {
    if (tid < st) red[tid] = fmaxf(red[tid], red[tid + st]);
    __syncthreads();
  }
  mx = red[0]; __syncthreads();

  float sm = 0.f;
  for (int s = tid; s < SEQ; s += 256) { float e = __expf(se[s] - mx); se[s] = e; sm += e; }
  red[tid] = sm; __syncthreads();
  for (int st = 128; st > 0; st >>= 1) {
    if (tid < st) red[tid] += red[tid + st];
    __syncthreads();
  }
  const float invZ = 1.0f / red[0];
  __syncthreads();

  const int d = tid & 63, stripe = tid >> 6;      // 4 s-stripes
  float acc = 0.f;
  for (int s = stripe; s < SEQ; s += 4)
    acc += se[s] * X[((size_t)b * SEQ + s) * HIDN + h * HDIM + d];
  pr[tid] = acc; __syncthreads();
  if (tid < 64) {
    float v = (pr[tid] + pr[tid + 64] + pr[tid + 128] + pr[tid + 192]) * invZ;
    if (gain) v *= gain[(size_t)b * HIDN + h * HDIM + tid];
    pooled[(size_t)b * HIDN + h * HDIM + tid] = v;
  }
}

// ---------------------------------------------------------------------------
extern "C" void kernel_launch(void* const* d_in, const int* in_sizes, int n_in,
                              void* d_out, int out_size, void* d_ws, size_t ws_size,
                              hipStream_t stream) {
  const float* hs   = (const float*)d_in[0];
  const float* mask = (const float*)d_in[1];
  const float* Wq   = (const float*)d_in[2];
  const float* bq   = (const float*)d_in[3];
  const float* Wqa  = (const float*)d_in[4];
  const float* bqa  = (const float*)d_in[5];
  const float* Wk   = (const float*)d_in[6];
  const float* bk   = (const float*)d_in[7];
  const float* Wka  = (const float*)d_in[8];
  const float* bka  = (const float*)d_in[9];
  const float* Wt   = (const float*)d_in[10];
  const float* bt   = (const float*)d_in[11];
  float* out = (float*)d_out;

  // workspace layout
  char* ws = (char*)d_ws;
  const size_t szMQ  = (size_t)MTOT * HIDN * sizeof(float);   // 128 MiB
  const size_t szMQb = (size_t)MTOT * HIDN * sizeof(__bf16);  //  64 MiB
  const size_t szSc  = (size_t)BATCH * NHEAD * SEQ * sizeof(float);
  float*  MQ       = (float*)ws;
  __bf16* MQb      = (__bf16*)(ws + szMQ);
  float*  scores   = (float*)(ws + szMQ + szMQb);
  float*  pooled_q = (float*)(ws + szMQ + szMQb + szSc);
  float*  pooled_k = pooled_q + (size_t)BATCH * HIDN;
  float*  MK       = out;   // d_out doubles as scratch for mixed_key; it is
                            // fully consumed before k_gemm_final overwrites it.

  dim3 gg(HIDN / 256, MTOT / 128);   // (4, 256)

  // 1) mixed_query (f32 + bf16 copy), 2) mixed_key (into d_out scratch)
  k_gemm_f32<true ><<<gg, 256, 0, stream>>>(hs, Wq, bq, MQ, MQb);
  k_gemm_f32<false><<<gg, 256, 0, stream>>>(hs, Wk, bk, MK, (__bf16*)nullptr);

  // 3) q scores -> 4) softmax + pooled_q
  k_score<<<MTOT / 8, 256, 0, stream>>>(MQ, Wqa, bqa, mask, (const float*)nullptr, scores);
  k_softpool<<<BATCH * NHEAD, 256, 0, stream>>>(scores, MQ, (const float*)nullptr, pooled_q);

  // 5) qk scores (gain = pooled_q) -> 6) softmax + pooled_k (gain = pooled_q)
  k_score<<<MTOT / 8, 256, 0, stream>>>(MK, Wka, bka, mask, pooled_q, scores);
  k_softpool<<<BATCH * NHEAD, 256, 0, stream>>>(scores, MK, pooled_q, pooled_k);

  // 7) out = MQ_bf16 @ (Wt ⊙ pooled_k_b)^T + bt + MQ
  k_gemm_final<<<gg, 256, 0, stream>>>(MQb, Wt, bt, pooled_k, MQ, out);
}
